// EdgeTypePredictor_66992899883306
// MI455X (gfx1250) — matrix operand order repaired and checked by reading, not compiled, not run on hardware
//
#include <hip/hip_runtime.h>
#include <hip/hip_bf16.h>

// ---------------------------------------------------------------------------
// EdgeTypePredictor on MI455X (gfx1250)
//   h = relu(concat(x[src],x[dst]) @ W0 + b0)    [E,512] x [512,512]
//   h = relu(h @ W1 + b1)                        [E,512] x [512,512]
//   h = h @ W2 + b2 ; pair-mean ; repeat         [E,512] x [512,5]
// Compute bound (~210 GFLOP vs ~20 MB unique data).  bf16 WMMA, fp32 acc.
// Weights staged per-workgroup into LDS with global_load_async_to_lds_b128
// (double buffered, ASYNCcnt); v_wmma consumes batched ds_load_b128 bursts.
// A/H tiles padded to 520 bf16/row (1040 B = 260 dwords, +4 mod 64 banks) so
// the 16 rows of an A-fragment load spread across all 64 LDS banks.
// ---------------------------------------------------------------------------

typedef __attribute__((ext_vector_type(16))) __bf16 v16bf;
typedef __attribute__((ext_vector_type(8)))  float  v8f;

#define IN_DIM   256
#define HID      512
#define OUT_DIM  5
#define TILE_M   128         // edges per block
#define THREADS  256         // 8 wave32s -> 8 x 16 rows = 128 rows
#define ROW_PITCH (HID + 8)  // 520 bf16 = 1040 B row pitch (bank-conflict pad)
#define NT_TOTAL (HID / 16)  // 32 N-tiles of 16 cols
#define KT_TOTAL (HID / 32)  // 16 K-steps of 32
#define KSTEP    2           // K-steps staged per barrier (16 KB)
#define NSTAGES  (KT_TOTAL / KSTEP)
#define STAGE_ELEMS (KSTEP * 8 * 512)  // bf16 elems per stage buffer (16 KB)

union Frag { uint4 q[2]; v16bf v; };

// ---------------------------------------------------------------------------
// Swizzle fp32 row-major W[K][N] into bf16 WMMA B-operand fragments.
// Fragment (kt,nt) covers K=kt*32..+31, N=nt*16..+15, stored as
// [lane 0..31][16 bf16]: lane = (kin>=16 ? 16 : 0) + col, the 16 bf16 are
// K = (lane/16)*16 + 0..15 ascending (contiguous k order, same as A) -> contiguous b128s.
// ---------------------------------------------------------------------------
__global__ __launch_bounds__(256) void swizzle_w_kernel(
    const float* __restrict__ W, __bf16* __restrict__ out, int K, int N) {
  int idx = blockIdx.x * blockDim.x + threadIdx.x;
  if (idx >= K * N) return;
  int k = idx / N, n = idx % N;
  int kTile = k >> 5, kin = k & 31;
  int nTile = n >> 4, col = n & 15;
  int lane  = ((kin >> 4) << 4) | col;
  int kk    = kin & 15;
  int nTiles = N >> 4;
  size_t o = (((size_t)(kTile * nTiles + nTile)) * 32 + lane) * 16 + kk;
  out[o] = (__bf16)W[idx];
}

// ---------------------------------------------------------------------------
// Async-copy one K-step's 8 fragments (8 KB) global -> LDS.  256 threads x
// 16 B x 2 issues.  GVS addressing: mem = SADDR + VADDR + IOFFSET, and the
// same IOFFSET is added to the LDS destination (ISA 08_async_tensor 4.4).
// ---------------------------------------------------------------------------
__device__ __forceinline__ void async_stage_b(const __bf16* __restrict__ wsw,
                                              int kt, int ntBase,
                                              __bf16* dstLds, int tid) {
  unsigned long long g =
      (unsigned long long)(uintptr_t)(wsw + (size_t)(kt * NT_TOTAL + ntBase) * 512);
  unsigned int lds  = (unsigned int)(uintptr_t)dstLds + (unsigned int)tid * 16u;
  unsigned int voff = (unsigned int)tid * 16u;
  asm volatile("global_load_async_to_lds_b128 %0, %1, %2"
               :: "v"(lds), "v"(voff), "s"(g) : "memory");
  asm volatile("global_load_async_to_lds_b128 %0, %1, %2 offset:4096"
               :: "v"(lds), "v"(voff), "s"(g) : "memory");
}

__device__ __forceinline__ void wait_async_barrier() {
  asm volatile("s_wait_asynccnt 0x0" ::: "memory");
  __syncthreads();
}

// ---------------------------------------------------------------------------
// One 512->512 layer for this block's 128-row LDS tile.  Wave w owns rows
// w*16..w*16+15.  A fragments: LDS, contiguous k order (lane g*16+m holds
// K = g*16+0..15) -> two ds_load_b128.  B fragments: staged
// LDS double buffer shared by all waves.  All 36 fragment loads of a stage
// are issued as one batch before the 16 WMMAs so DScnt waits amortize.
// ---------------------------------------------------------------------------
__device__ __forceinline__ void mlp_layer512(
    const __bf16* sIn, __bf16* sOut, __bf16* sB,
    const __bf16* __restrict__ wsw, const float* __restrict__ bias,
    int wave, int lane, int tid) {
  const int g  = lane >> 4;
  const int mr = lane & 15;
  const int m0 = wave * 16;
  const __bf16* aRow = sIn + (size_t)(m0 + mr) * ROW_PITCH;

  for (int nBase = 0; nBase < HID; nBase += 128) {  // 8 N-tiles per chunk
    const int ntBase = nBase >> 4;
    v8f acc[8];
    const v8f vz = {0.f, 0.f, 0.f, 0.f, 0.f, 0.f, 0.f, 0.f};
#pragma unroll
    for (int i = 0; i < 8; ++i) acc[i] = vz;

    // prologue: stage 0
#pragma unroll
    for (int kk = 0; kk < KSTEP; ++kk)
      async_stage_b(wsw, kk, ntBase, sB + kk * (8 * 512), tid);
    wait_async_barrier();

    for (int ks = 0; ks < NSTAGES; ++ks) {
      __bf16* cur = sB + (ks & 1) * STAGE_ELEMS;
      if (ks + 1 < NSTAGES) {   // issue next stage, overlap with WMMA below
        __bf16* nxt = sB + ((ks + 1) & 1) * STAGE_ELEMS;
#pragma unroll
        for (int kk = 0; kk < KSTEP; ++kk)
          async_stage_b(wsw, (ks + 1) * KSTEP + kk, ntBase,
                        nxt + kk * (8 * 512), tid);
      }

      // ---- batch all fragment loads of this stage ----
      Frag a[KSTEP];
      Frag b[KSTEP][8];
#pragma unroll
      for (int kk = 0; kk < KSTEP; ++kk) {
        const int kt = ks * KSTEP + kk;
        const uint4* ap = (const uint4*)(aRow + kt * 32 + g * 16);
        a[kk].q[0] = ap[0];
        a[kk].q[1] = ap[1];  // +8 bf16 = +16 bytes
        const __bf16* bLane = cur + kk * (8 * 512) + lane * 16;
#pragma unroll
        for (int nt = 0; nt < 8; ++nt) {
          const uint4* bp = (const uint4*)(bLane + nt * 512);
          b[kk][nt].q[0] = bp[0];
          b[kk][nt].q[1] = bp[1];
        }
      }
      // ---- WMMA burst ----
#pragma unroll
      for (int kk = 0; kk < KSTEP; ++kk)
#pragma unroll
        for (int nt = 0; nt < 8; ++nt)
          acc[nt] = __builtin_amdgcn_wmma_f32_16x16x32_bf16(
              false, a[kk].v, false, b[kk][nt].v, (short)0, acc[nt],
              false, false);

      wait_async_barrier();  // next buffer staged; all waves done with cur
    }

    // bias + relu epilogue; D layout: VGPR i -> row m0 + g*8 + i, col = lane%16
#pragma unroll
    for (int nt = 0; nt < 8; ++nt) {
      int col = nBase + nt * 16 + mr;
      float bc = bias[col];
#pragma unroll
      for (int i = 0; i < 8; ++i) {
        int row = m0 + g * 8 + i;
        float v = acc[nt][i] + bc;
        v = v > 0.f ? v : 0.f;
        sOut[(size_t)row * ROW_PITCH + col] = (__bf16)v;
      }
    }
  }
}

__global__ __launch_bounds__(THREADS, 1) void edge_mlp_kernel(
    const float* __restrict__ x, const int* __restrict__ ei,
    const __bf16* __restrict__ w0sw, const float* __restrict__ b0,
    const __bf16* __restrict__ w1sw, const float* __restrict__ b1,
    const float* __restrict__ W2, const float* __restrict__ b2,
    float* __restrict__ out, int nEdges, int nNodes) {
  __shared__ __bf16 sA[TILE_M * ROW_PITCH];  // 130 KB: concat tile, then H1
  __shared__ __bf16 sH[TILE_M * ROW_PITCH];  // 130 KB: H0
  __shared__ __bf16 sB[2 * STAGE_ELEMS];     //  32 KB: weight stage buffers
  __shared__ float  sW2[HID * OUT_DIM];      //  10 KB: layer-2 weights

  const int tid  = threadIdx.x;
  const int lane = tid & 31;
  const int wave = tid >> 5;
  const int e0   = blockIdx.x * TILE_M;

  // ---- gather endpoint rows, fp32 -> bf16, into LDS concat tile ----
  {
    int eloc = tid >> 1;       // 128 rows
    int half = tid & 1;        // src / dst half of concat
    int eg = e0 + eloc;
    int node = (eg < nEdges) ? ei[half * nEdges + eg] : 0;
    if (node < 0) node = 0;
    if (node >= nNodes) node = nNodes - 1;
    const float4* src = (const float4*)(x + (size_t)node * IN_DIM);
    __bf16* dst = sA + (size_t)eloc * ROW_PITCH + half * IN_DIM;
#pragma unroll 4
    for (int i = 0; i < IN_DIM / 4; ++i) {
      float4 f = src[i];
      dst[i * 4 + 0] = (__bf16)f.x;
      dst[i * 4 + 1] = (__bf16)f.y;
      dst[i * 4 + 2] = (__bf16)f.z;
      dst[i * 4 + 3] = (__bf16)f.w;
    }
  }
  // ---- preload W2 into LDS (used after the two WMMA layers) ----
  for (int i = tid; i < HID * OUT_DIM; i += THREADS) sW2[i] = W2[i];
  __syncthreads();

  mlp_layer512(sA, sH, sB, w0sw, b0, wave, lane, tid);  // H0
  __syncthreads();
  mlp_layer512(sH, sA, sB, w1sw, b1, wave, lane, tid);  // H1 -> sA
  __syncthreads();

  // ---- layer 2 (512 -> 5, ~1% of FLOPs) + pair symmetrization, VALU ----
  for (int i = tid; i < (TILE_M / 2) * OUT_DIM; i += THREADS) {
    int p = i / OUT_DIM, o = i % OUT_DIM;
    int ea = p * 2, eb = ea + 1;
    const __bf16* ha = sA + (size_t)ea * ROW_PITCH;
    const __bf16* hb = sA + (size_t)eb * ROW_PITCH;
    float s = 0.f;
    for (int k = 0; k < HID; ++k) {
      float w = sW2[k * OUT_DIM + o];
      s += ((float)ha[k] + (float)hb[k]) * w;
    }
    s = 0.5f * s + b2[o];  // mean over (i,j)/(j,i); bias commutes with mean
    int ego = e0 + ea;
    if (ego + 1 < nEdges) {
      out[(size_t)ego * OUT_DIM + o] = s;
      out[(size_t)(ego + 1) * OUT_DIM + o] = s;
    }
  }
}

extern "C" void kernel_launch(void* const* d_in, const int* in_sizes, int n_in,
                              void* d_out, int out_size, void* d_ws, size_t ws_size,
                              hipStream_t stream) {
  const float* x  = (const float*)d_in[0];
  const int*   ei = (const int*)d_in[1];   // dense_edge_index, [2, E]
  const float* W0 = (const float*)d_in[2];
  const float* b0 = (const float*)d_in[3];
  const float* W1 = (const float*)d_in[4];
  const float* b1 = (const float*)d_in[5];
  const float* W2 = (const float*)d_in[6];
  const float* b2 = (const float*)d_in[7];
  float* out = (float*)d_out;

  const int nNodes = in_sizes[0] / IN_DIM;
  const int nEdges = in_sizes[1] / 2;

  // workspace: swizzled bf16 weights (512 KB + 512 KB)
  __bf16* w0sw = (__bf16*)d_ws;
  __bf16* w1sw = w0sw + (size_t)(2 * IN_DIM) * HID;

  {
    int total = (2 * IN_DIM) * HID;
    swizzle_w_kernel<<<(total + 255) / 256, 256, 0, stream>>>(W0, w0sw,
                                                              2 * IN_DIM, HID);
  }
  {
    int total = HID * HID;
    swizzle_w_kernel<<<(total + 255) / 256, 256, 0, stream>>>(W1, w1sw,
                                                              HID, HID);
  }

  int blocks = (nEdges + TILE_M - 1) / TILE_M;  // 1563
  edge_mlp_kernel<<<blocks, THREADS, 0, stream>>>(x, ei, w0sw, b0, w1sw, b1,
                                                  W2, b2, out, nEdges, nNodes);
}
